// REGATConv_52656299049564
// MI455X (gfx1250) — matrix operand order repaired and checked by reading, not compiled, not run on hardware
//
#include <hip/hip_runtime.h>
#include <hip/hip_bf16.h>

#define NH    4      // heads
#define NDH   64     // out feats per head
#define NHD   256    // NH * NDH
#define KIN   256    // in feats
#define ALPHA_C 100.0f

typedef __attribute__((ext_vector_type(16))) __bf16 v16bf;
typedef __attribute__((ext_vector_type(8)))  __bf16 v8bf;
typedef __attribute__((ext_vector_type(8)))  float  v8f;

__device__ __forceinline__ __bf16 f2bf(float f) {
    unsigned u = __float_as_uint(f);
    u += 0x7FFFu + ((u >> 16) & 1u);          // round-to-nearest-even
    unsigned short s = (unsigned short)(u >> 16);
    return __builtin_bit_cast(__bf16, s);
}
__device__ __forceinline__ float lrelu(float x, float s) { return x >= 0.0f ? x : s * x; }
// monotone float <-> uint key for atomicMax-based segment max
__device__ __forceinline__ unsigned fkey(float f) {
    unsigned b = __float_as_uint(f);
    return (b & 0x80000000u) ? ~b : (b | 0x80000000u);
}
__device__ __forceinline__ float kdec(unsigned k) {
    return __uint_as_float((k & 0x80000000u) ? (k & 0x7FFFFFFFu) : ~k);
}
__device__ __forceinline__ void atomAddF(float* p, float v) {
    __hip_atomic_fetch_add(p, v, __ATOMIC_RELAXED, __HIP_MEMORY_SCOPE_AGENT);
}

// ---- 1) fp32 -> bf16 for feat; fp32 -> transposed bf16 for fc_w -------------
__global__ void convert_kernel(const float* __restrict__ feat,
                               const float* __restrict__ fcw,
                               __bf16* __restrict__ featbf,
                               __bf16* __restrict__ fcwt,   // [KIN][NHD]: fcwt[k*NHD+o]=fcw[o*KIN+k]
                               long long nf) {
    long long i = (long long)blockIdx.x * blockDim.x + threadIdx.x;
    if (i < nf) {
        featbf[i] = f2bf(feat[i]);
    } else {
        long long j = i - nf;
        if (j < (long long)KIN * NHD) {
            int k = (int)(j >> 8);
            int o = (int)(j & 255);
            fcwt[j] = f2bf(fcw[(long long)o * KIN + k]);
        }
    }
}

// ---- 2) zero output / denom / emax keys ------------------------------------
__global__ void init_kernel(float* __restrict__ out, float* __restrict__ denom,
                            unsigned* __restrict__ emaxk, long long n_out, long long n_nh) {
    long long i = (long long)blockIdx.x * blockDim.x + threadIdx.x;
    if (i < n_out) out[i] = 0.0f;
    if (i < n_nh) { denom[i] = 0.0f; emaxk[i] = 0u; }
}

// ---- 3) WMMA GEMM: ft[N,256] = feat_bf16[N,256] @ fcwt_bf16[256,256] --------
// block = 512 threads = 16 waves; wave w computes 16x16 tile (m0, n0 = 16*w)
__global__ void gemm_wmma_kernel(const __bf16* __restrict__ A,   // [N, KIN]
                                 const __bf16* __restrict__ Bt,  // [KIN, NHD] (k-major)
                                 float* __restrict__ C,          // [N, NHD]
                                 int n) {
    const int lane = threadIdx.x & 31;
    const int wave = threadIdx.x >> 5;        // 0..15 -> column tile
    const int r    = lane & 15;
    const int hs   = lane >> 4;               // lane half
    const int m0   = blockIdx.x * 16;
    const int n0   = wave * 16;

    int rowA = m0 + r;
    if (rowA >= n) rowA = n - 1;              // clamp (N is a multiple of 16 anyway)
    const __bf16* arow = A + (size_t)rowA * KIN;

    v8f acc = {};
    #pragma unroll
    for (int k0 = 0; k0 < KIN; k0 += 32) {
        // A: lane=M, regs=K; lanes 0-15 take K {k0..k0+7, k0+16..k0+23},
        //                     lanes 16-31 take K {k0+8..k0+15, k0+24..k0+31}
        v8bf alo = *(const v8bf*)(arow + k0 + 8 * hs);
        v8bf ahi = *(const v8bf*)(arow + k0 + 16 + 8 * hs);
        v16bf a = __builtin_shufflevector(alo, ahi,
                    0, 1, 2, 3, 4, 5, 6, 7, 8, 9, 10, 11, 12, 13, 14, 15);
        // B: lane=N (n0+r), regs=K in the same order as A
        v16bf b;
        #pragma unroll
        for (int s = 0; s < 16; ++s) {
            int kk = k0 + (s < 8 ? 8 * hs + s : 16 + 8 * hs + (s - 8));
            b[s] = Bt[(size_t)kk * NHD + n0 + r];
        }
        acc = __builtin_amdgcn_wmma_f32_16x16x32_bf16(
                  false, a, false, b, (short)0, acc, false, false);
    }
    // D: lane=N (n0+r), regs=M (v + 8*hs)
    #pragma unroll
    for (int v = 0; v < 8; ++v) {
        int row = m0 + v + 8 * hs;
        if (row < n) C[(size_t)row * NHD + n0 + r] = acc[v];
    }
}

// ---- 4) el/er: [N,NH] attention projections ---------------------------------
__global__ void eler_kernel(const float* __restrict__ ft,
                            const float* __restrict__ attn_l,
                            const float* __restrict__ attn_r,
                            float* __restrict__ el, float* __restrict__ er, int n) {
    __shared__ float sl[NHD];
    __shared__ float sr[NHD];
    int node = blockIdx.x;
    int t = threadIdx.x;               // t = h*64 + d
    float v = ft[(size_t)node * NHD + t];
    sl[t] = v * attn_l[t];
    sr[t] = v * attn_r[t];
    __syncthreads();
    #pragma unroll
    for (int s = 32; s > 0; s >>= 1) {
        if ((t & 63) < s) { sl[t] += sl[t + s]; sr[t] += sr[t + s]; }
        __syncthreads();
    }
    if ((t & 63) == 0) {
        int h = t >> 6;
        el[(size_t)node * NH + h] = sl[t];
        er[(size_t)node * NH + h] = sr[t];
    }
    (void)n;
}

// ---- 5) per-edge logits + segment max (atomic uint key) ---------------------
__global__ void edge_logit_kernel(const int* __restrict__ src, const int* __restrict__ dst,
                                  const int* __restrict__ etype,
                                  const float* __restrict__ edge_w,   // [8, NH]
                                  const float* __restrict__ el, const float* __restrict__ er,
                                  float* __restrict__ ebuf, unsigned* __restrict__ emaxk,
                                  int e) {
    int idx = blockIdx.x * blockDim.x + threadIdx.x;
    if (idx >= e * NH) return;
    int eid = idx >> 2, h = idx & 3;
    int s = src[eid], d = dst[eid], et = etype[eid] - 1;
    float ew = lrelu(edge_w[et * NH + h] * ALPHA_C, 0.01f);
    float x = lrelu((el[(size_t)s * NH + h] + er[(size_t)d * NH + h]) * ew, 0.2f);
    ebuf[idx] = x;
    atomicMax(&emaxk[(size_t)d * NH + h], fkey(x));
}

// ---- 6) exp(e - max) and segment sum ---------------------------------------
__global__ void edge_exp_kernel(const int* __restrict__ dst,
                                float* __restrict__ ebuf,
                                const unsigned* __restrict__ emaxk,
                                float* __restrict__ denom, int e) {
    int idx = blockIdx.x * blockDim.x + threadIdx.x;
    if (idx >= e * NH) return;
    int eid = idx >> 2, h = idx & 3;
    int d = dst[eid];
    float mx = kdec(emaxk[(size_t)d * NH + h]);
    float ex = __expf(ebuf[idx] - mx);
    ebuf[idx] = ex;
    atomAddF(&denom[(size_t)d * NH + h], ex);
}

// ---- 7) normalize + weighted scatter-sum ------------------------------------
// one block (256 threads) per edge; thread t handles feature t (h = t>>6)
__global__ void scatter_kernel(const int* __restrict__ src, const int* __restrict__ dst,
                               const float* __restrict__ ebuf,
                               const float* __restrict__ denom,
                               const float* __restrict__ ft,
                               float* __restrict__ out, int e) {
    int eid = blockIdx.x;
    if (eid >= e) return;
    int t = threadIdx.x;
    int h = t >> 6;
    int s = src[eid], d = dst[eid];
    float a = ebuf[(size_t)eid * NH + h] / denom[(size_t)d * NH + h];
    float msg = a * ft[(size_t)s * NHD + t];
    atomAddF(&out[(size_t)d * NHD + t], msg);
}

extern "C" void kernel_launch(void* const* d_in, const int* in_sizes, int n_in,
                              void* d_out, int out_size, void* d_ws, size_t ws_size,
                              hipStream_t stream) {
    const float* feat   = (const float*)d_in[0];
    const int*   src    = (const int*)d_in[1];
    const int*   dst    = (const int*)d_in[2];
    const int*   etype  = (const int*)d_in[3];
    const float* fcw    = (const float*)d_in[4];
    const float* attn_l = (const float*)d_in[5];
    const float* attn_r = (const float*)d_in[6];
    const float* edge_w = (const float*)d_in[7];
    float* out = (float*)d_out;

    const int n = in_sizes[0] / KIN;   // 50000
    const int e = in_sizes[1];         // 800000

    // workspace carve-up (256B aligned)
    char* base = (char*)d_ws;
    size_t off = 0;
    auto carve = [&](size_t bytes) -> char* {
        char* p = base + off;
        off = (off + bytes + 255) & ~(size_t)255;
        return p;
    };
    float*    ft     = (float*)   carve((size_t)n * NHD * sizeof(float));
    __bf16*   featbf = (__bf16*)  carve((size_t)n * KIN * sizeof(__bf16));
    __bf16*   fcwt   = (__bf16*)  carve((size_t)KIN * NHD * sizeof(__bf16));
    float*    el     = (float*)   carve((size_t)n * NH * sizeof(float));
    float*    er     = (float*)   carve((size_t)n * NH * sizeof(float));
    unsigned* emaxk  = (unsigned*)carve((size_t)n * NH * sizeof(unsigned));
    float*    denom  = (float*)   carve((size_t)n * NH * sizeof(float));
    float*    ebuf   = (float*)   carve((size_t)e * NH * sizeof(float));
    (void)ws_size; (void)n_in; (void)out_size;

    const long long nf = (long long)n * KIN;

    // 1) precision convert + weight transpose
    {
        long long total = nf + (long long)KIN * NHD;
        int blocks = (int)((total + 255) / 256);
        convert_kernel<<<blocks, 256, 0, stream>>>(feat, fcw, featbf, fcwt, nf);
    }
    // 2) init accumulators
    {
        long long n_out = (long long)n * NHD;
        long long n_nh  = (long long)n * NH;
        int blocks = (int)((n_out + 255) / 256);
        init_kernel<<<blocks, 256, 0, stream>>>(out, denom, emaxk, n_out, n_nh);
    }
    // 3) WMMA GEMM: ft = feat @ fc_w.T   (16 waves per block, one 16-row strip)
    {
        int blocks = (n + 15) / 16;   // 3125
        gemm_wmma_kernel<<<blocks, 512, 0, stream>>>(featbf, fcwt, ft, n);
    }
    // 4) el / er
    eler_kernel<<<n, NHD, 0, stream>>>(ft, attn_l, attn_r, el, er, n);
    // 5) edge logits + segment max
    {
        int blocks = (e * NH + 255) / 256;
        edge_logit_kernel<<<blocks, 256, 0, stream>>>(src, dst, etype, edge_w,
                                                      el, er, ebuf, emaxk, e);
    }
    // 6) exp + segment sum
    {
        int blocks = (e * NH + 255) / 256;
        edge_exp_kernel<<<blocks, 256, 0, stream>>>(dst, ebuf, emaxk, denom, e);
    }
    // 7) normalize + scatter-sum into output
    scatter_kernel<<<e, NHD, 0, stream>>>(src, dst, ebuf, denom, ft, out, e);
}
